// TopKRouter_2877628088600
// MI455X (gfx1250) — matrix operand order repaired and checked by reading, not compiled, not run on hardware
//
#include <hip/hip_runtime.h>
#include <math.h>

typedef __attribute__((ext_vector_type(16))) __bf16 v16bf;
typedef __attribute__((ext_vector_type(8)))  float  v8f;
typedef __attribute__((ext_vector_type(4)))  float  f4;

#define T_TOKENS 131072
#define D_IN     1024
#define NE       64
#define A_TILES      2                                // 32 tokens per wave
#define TOK_PER_WAVE (16 * A_TILES)
#define WAVES_PER_WG 8
#define TOK_PER_WG   (TOK_PER_WAVE * WAVES_PER_WG)    // 256
#define LDS_BYTES    (2u * NE * D_IN * 2u)            // hi+lo bf16 = 262144

__global__ __launch_bounds__(256) void topk_router_kernel(
    const float* __restrict__ x, const float* __restrict__ W,
    int* __restrict__ out_idx, float* __restrict__ out_w)
{
    extern __shared__ char smem[];
    __bf16* ldsWhi = (__bf16*)smem;              // [NE][D_IN] bf16 hi
    __bf16* ldsWlo = ldsWhi + NE * D_IN;         // [NE][D_IN] bf16 lo (residual)

    const int tid = threadIdx.x;

    // ---- one-time: convert W fp32 -> bf16 hi/lo into LDS ----
    {
        const f4* W4 = (const f4*)W;
        for (int v = tid; v < (NE * D_IN) / 4; v += 256) {
            f4 w = W4[v];
#pragma unroll
            for (int e = 0; e < 4; ++e) {
                float  f = w[e];
                __bf16 h = (__bf16)f;
                ldsWhi[4 * v + e] = h;
                ldsWlo[4 * v + e] = (__bf16)(f - (float)h);
            }
        }
    }
    __syncthreads();

    const int lane   = tid & 31;
    const int wv     = tid >> 5;
    const int laneM  = lane & 15;     // A row / B column within tile
    const int khalf  = lane >> 4;     // which K half this lane holds
    const int token_base = blockIdx.x * TOK_PER_WG + wv * TOK_PER_WAVE;

    const float* xrow[A_TILES];
#pragma unroll
    for (int a = 0; a < A_TILES; ++a)
        xrow[a] = x + (size_t)(token_base + a * 16 + laneM) * D_IN;

    v8f c[A_TILES][4] = {};   // 32 tokens x 64 experts (8 accumulator tiles)

#pragma unroll 1
    for (int k0 = 0; k0 < D_IN; k0 += 32) {
        if (k0 + 64 < D_IN) {
#pragma unroll
            for (int a = 0; a < A_TILES; ++a)
                __builtin_prefetch(xrow[a] + k0 + 64, 0, 3);  // global_prefetch_b8
        }

        // ---- B operands: all 4 expert tiles, hi+lo (ds_load_b128 pairs) ----
        const int koff = k0 + khalf * 16;
        v16bf bhi[4], blo[4];
#pragma unroll
        for (int t = 0; t < 4; ++t) {
            const int e = t * 16 + laneM;  // expert (B column)
            bhi[t] = *(const v16bf*)(ldsWhi + (size_t)e * D_IN + koff);
            blo[t] = *(const v16bf*)(ldsWlo + (size_t)e * D_IN + koff);
        }

        // ---- A operands: 2 token tiles, 16 fp32 each, split into bf16 hi/lo
        // lanes 0-15 hold K = k0+{0..15}; lanes 16-31 hold K = k0+{16..31}
        v16bf ahi[A_TILES], alo[A_TILES];
#pragma unroll
        for (int a = 0; a < A_TILES; ++a) {
            const f4* pa = (const f4*)(xrow[a] + k0 + khalf * 16);
            f4 q0 = __builtin_nontemporal_load(pa);
            f4 q1 = __builtin_nontemporal_load(pa + 1);
            const f4* pb = (const f4*)(xrow[a] + k0 + khalf * 16 + 8);
            f4 q2 = __builtin_nontemporal_load(pb);
            f4 q3 = __builtin_nontemporal_load(pb + 1);

            float av[16];
#pragma unroll
            for (int e = 0; e < 4; ++e) {
                av[e]      = q0[e];
                av[4 + e]  = q1[e];
                av[8 + e]  = q2[e];
                av[12 + e] = q3[e];
            }
#pragma unroll
            for (int i = 0; i < 16; ++i) {
                __bf16 h = (__bf16)av[i];
                ahi[a][i] = h;
                alo[a][i] = (__bf16)(av[i] - (float)h);
            }
        }

        // ---- fp32-accurate split-GEMM: xh*Wh + xl*Wh + xh*Wl ----
        // Term-major over 8 independent accumulators: same-accumulator WMMAs
        // are separated by 7 independent ones (> 5-slot bf16 XDL hazard).
#pragma unroll
        for (int t = 0; t < 4; ++t)
#pragma unroll
            for (int a = 0; a < A_TILES; ++a)
                c[a][t] = __builtin_amdgcn_wmma_f32_16x16x32_bf16(
                              false, ahi[a], false, bhi[t], (short)0, c[a][t], false, false);
#pragma unroll
        for (int t = 0; t < 4; ++t)
#pragma unroll
            for (int a = 0; a < A_TILES; ++a)
                c[a][t] = __builtin_amdgcn_wmma_f32_16x16x32_bf16(
                              false, alo[a], false, bhi[t], (short)0, c[a][t], false, false);
#pragma unroll
        for (int t = 0; t < 4; ++t)
#pragma unroll
            for (int a = 0; a < A_TILES; ++a)
                c[a][t] = __builtin_amdgcn_wmma_f32_16x16x32_bf16(
                              false, ahi[a], false, blo[t], (short)0, c[a][t], false, false);
    }

    // ---- top-2 + softmax directly from C layout ----
    // Tile (a,t): VGPR r, lanes 0-15 -> token (base+a*16+r), lanes 16-31 -> +8,
    //             expert = t*16 + (lane&15)
#pragma unroll 1
    for (int a = 0; a < A_TILES; ++a) {
#pragma unroll 1
        for (int r = 0; r < 8; ++r) {
            float b = -INFINITY, s = -INFINITY;
            int   ib = 0x7fffffff, is = 0x7fffffff;

            auto cand = [&](float v, int i) {
                if (v > b || (v == b && i < ib)) { s = b; is = ib; b = v; ib = i; }
                else if (v > s || (v == s && i < is)) { s = v; is = i; }
            };

#pragma unroll
            for (int t = 0; t < 4; ++t)
                cand(c[a][t][r], t * 16 + laneM);

            // butterfly merge across the 16-lane half (bit4 never touched)
#pragma unroll
            for (int m = 1; m < 16; m <<= 1) {
                float ob  = __shfl_xor(b,  m, 32);
                int   oib = __shfl_xor(ib, m, 32);
                float os  = __shfl_xor(s,  m, 32);
                int   ois = __shfl_xor(is, m, 32);
                cand(ob, oib);
                cand(os, ois);
            }

            if (laneM == 0) {
                const int tok = token_base + a * 16 + r + khalf * 8;
                const float e1 = __expf(s - b);          // stable 2-way softmax
                const float inv = 1.0f / (1.0f + e1);
                out_idx[tok * 2 + 0] = ib;
                out_idx[tok * 2 + 1] = is;
                out_w[tok * 2 + 0]   = inv;
                out_w[tok * 2 + 1]   = e1 * inv;
            }
        }
    }
}

extern "C" void kernel_launch(void* const* d_in, const int* in_sizes, int n_in,
                              void* d_out, int out_size, void* d_ws, size_t ws_size,
                              hipStream_t stream) {
    (void)in_sizes; (void)n_in; (void)d_ws; (void)ws_size; (void)out_size;
    const float* x = (const float*)d_in[0];
    const float* W = (const float*)d_in[1];
    // d_in[2] is top_k (==2), compile-time specialized.

    int*   out_idx = (int*)d_out;                          // [T,2] int32 first
    float* out_w   = (float*)d_out + (size_t)T_TOKENS * 2; // then [T,2] fp32

    (void)hipFuncSetAttribute((const void*)topk_router_kernel,
                              hipFuncAttributeMaxDynamicSharedMemorySize,
                              (int)LDS_BYTES);

    topk_router_kernel<<<T_TOKENS / TOK_PER_WG, 256, LDS_BYTES, stream>>>(
        x, W, out_idx, out_w);
}
